// ScaledDotProductAttention_10788957847895
// MI455X (gfx1250) — matrix operand order repaired and checked by reading, not compiled, not run on hardware
//
#include <hip/hip_runtime.h>

// Shapes from the reference
#define B_  16
#define L_  2048
#define D_  64
#define SCALE_ 0.125f        // 1 / TEMPERATURE (= 1/sqrt(64))
#define SROW  2052           // padded f32-dword stride of score rows (anti bank-conflict)

typedef __attribute__((ext_vector_type(16))) __bf16       v16bf;
typedef __attribute__((ext_vector_type(8)))  float        v8f;
typedef __attribute__((ext_vector_type(4)))  float        f4v;
typedef __attribute__((ext_vector_type(4)))  unsigned int u4v;
typedef __attribute__((ext_vector_type(2)))  unsigned int u2v;

union ABFrag { v16bf v; unsigned int u[8]; u4v q[2]; };

// f32 pair -> packed bf16 dword in ONE v_perm_b32 (truncating round)
__device__ __forceinline__ unsigned int pack_bf16(float lo, float hi) {
    return __builtin_amdgcn_perm(__float_as_uint(hi), __float_as_uint(lo),
                                 0x07060302u);
}

// ---------- preprocessing: K f32 -> packed bf16 d-pairs, row-major ----------
__global__ __launch_bounds__(256)
void pack_k_kernel(const float* __restrict__ K, unsigned int* __restrict__ Kbf) {
    size_t i = (size_t)blockIdx.x * 256 + threadIdx.x;   // over B_*L_*(D_/2)
    float2 f = ((const float2*)K)[i];
    Kbf[i] = pack_bf16(f.x, f.y);
}

// ---------- preprocessing: V f32 -> bf16 row-pair transpose Vsw[b][n][rp] ----------
// Vsw dword (b, n, rp) = pack(V[b][2rp][n], V[b][2rp+1][n])  -> exact WMMA-B fragment dwords
__global__ __launch_bounds__(128)
void pack_v_kernel(const float* __restrict__ V, unsigned int* __restrict__ Vsw) {
    __shared__ float sv[32][D_ + 1];
    const int b    = blockIdx.y;
    const int tile = blockIdx.x;            // 64 tiles of 32 rows
    const int tid  = threadIdx.x;
    const int r0   = tile * 32;

    const int rowt = tid >> 4, col4 = (tid & 15) * 4;
    #pragma unroll
    for (int j = 0; j < 4; ++j) {
        int row = rowt + 8 * j;
        f4v f = *(const f4v*)(V + ((size_t)b * L_ + r0 + row) * D_ + col4);
        sv[row][col4 + 0] = f.x;  sv[row][col4 + 1] = f.y;
        sv[row][col4 + 2] = f.z;  sv[row][col4 + 3] = f.w;
    }
    __syncthreads();

    const int n    = tid >> 1;
    const int rpl0 = (tid & 1) * 8;
    unsigned int* outp = Vsw + ((size_t)b * D_ + n) * (L_ / 2) + (size_t)tile * 16 + rpl0;
    #pragma unroll
    for (int j = 0; j < 8; ++j) {
        int rp = rpl0 + j;
        outp[j] = pack_bf16(sv[2 * rp][n], sv[2 * rp + 1][n]);
    }
}

// ---------- main fused attention kernel: 256 threads = 8 waves ----------
__global__ __launch_bounds__(256)
void sdpa_wmma_kernel(const float* __restrict__ Q,
                      const unsigned int* __restrict__ Kbf,
                      const unsigned int* __restrict__ Vsw,
                      const unsigned char* __restrict__ M,
                      float* __restrict__ Out,
                      float* __restrict__ Attn)
{
    extern __shared__ float smem[];
    float* sS   = smem;                 // [16][SROW] f32 scores, later packed-bf16 P in-place
    float* sAcc = smem + 16 * SROW;     // [8][128] split-K partial accumulators

    const int b    = blockIdx.y;
    const int q0   = blockIdx.x * 16;
    const int tid  = threadIdx.x;
    const int wave = tid >> 5;
    const int lane = tid & 31;
    const int hh   = lane >> 4;         // lane half (0/1)
    const int l15  = lane & 15;

    // ---------------- Phase 1: S = (Q*scale) K^T ----------------
    ABFrag aq[2];
    {
        const float* qrow = Q + ((size_t)b * L_ + q0 + l15) * D_;
        #pragma unroll
        for (int w = 0; w < 2; ++w) {
            #pragma unroll
            for (int vr = 0; vr < 8; ++vr) {
                // A layout: lanes0-15: V0-7 K=0..15; lanes16-31 K=16..31
                int kb = 2 * vr + 16 * hh + 32 * w;
                float2 f = *(const float2*)(qrow + kb);
                aq[w].u[vr] = pack_bf16(f.x * SCALE_, f.y * SCALE_);  // fold 1/temp
            }
        }
    }

    for (int kt = wave; kt < L_ / 16; kt += 8) {
        const u4v* kb128 = (const u4v*)(Kbf +
            ((size_t)b * L_ + (size_t)kt * 16 + l15) * (D_ / 2) + 8 * hh);
        __builtin_prefetch((const char*)kb128 + (size_t)128 * (D_ / 2) * 4, 0, 3);
        ABFrag bk[2];
        bk[0].q[0] = kb128[0];  bk[0].q[1] = kb128[1];
        bk[1].q[0] = kb128[4];  bk[1].q[1] = kb128[5];

        v8f acc = {};
        acc = __builtin_amdgcn_wmma_f32_16x16x32_bf16(false, aq[0].v, false, bk[0].v,
                                                      (short)0, acc, false, false);
        acc = __builtin_amdgcn_wmma_f32_16x16x32_bf16(false, aq[1].v, false, bk[1].v,
                                                      (short)0, acc, false, false);
        #pragma unroll
        for (int r = 0; r < 8; ++r) {
            int m = r + 8 * hh;
            sS[m * SROW + kt * 16 + l15] = acc[r];
        }
    }
    __syncthreads();

    // ---------------- Phase 2: row softmax + attn writeout + in-place bf16 pack ----------------
    #pragma unroll
    for (int rr = 0; rr < 2; ++rr) {
        int r = wave * 2 + rr;
        const unsigned int* mrow32 =
            (const unsigned int*)(M + ((size_t)b * L_ + q0 + r) * L_);
        f4v* s4 = (f4v*)(sS + r * SROW);

        float mx = -3.0e38f;
        for (int i4 = lane; i4 < L_ / 4; i4 += 32) {
            f4v x = s4[i4];
            unsigned int mw = __builtin_nontemporal_load(mrow32 + i4);
            if (mw & 0x000000FFu) x.x = -3.0e38f;
            if (mw & 0x0000FF00u) x.y = -3.0e38f;
            if (mw & 0x00FF0000u) x.z = -3.0e38f;
            if (mw & 0xFF000000u) x.w = -3.0e38f;
            s4[i4] = x;
            mx = fmaxf(mx, fmaxf(fmaxf(x.x, x.y), fmaxf(x.z, x.w)));
        }
        #pragma unroll
        for (int o = 16; o > 0; o >>= 1) mx = fmaxf(mx, __shfl_xor(mx, o, 32));

        float s = 0.f;
        for (int i4 = lane; i4 < L_ / 4; i4 += 32) {
            f4v x = s4[i4], e;
            e.x = __expf(x.x - mx);  e.y = __expf(x.y - mx);
            e.z = __expf(x.z - mx);  e.w = __expf(x.w - mx);
            s4[i4] = e;
            s += (e.x + e.y) + (e.z + e.w);
        }
        #pragma unroll
        for (int o = 16; o > 0; o >>= 1) s += __shfl_xor(s, o, 32);

        float inv = 1.0f / s;
        f4v* arow = (f4v*)(Attn + ((size_t)b * L_ + q0 + r) * L_);
        unsigned int* prow = (unsigned int*)(sS + r * SROW);   // pack in place
        for (int i4 = lane; i4 < L_ / 4; i4 += 32) {
            f4v p = s4[i4] * inv;
            __builtin_nontemporal_store(p, arow + i4);   // 268MB stream: keep out of caches
            u2v pk;
            pk.x = pack_bf16(p.x, p.y);
            pk.y = pack_bf16(p.z, p.w);
            // write dwords [2*i4, 2*i4+1]: strictly behind every future read window
            *(u2v*)(prow + 2 * i4) = pk;
        }
    }
    __syncthreads();

    // ---------------- Phase 3: O = P V, split-K across wave pairs ----------------
    const int t    = wave >> 1;          // column tile 0..3 (16 dims each)
    const int halfk = wave & 1;          // kc range half
    const int dcol = t * 16;
    const u4v* pa = (const u4v*)((const unsigned int*)sS + (size_t)l15 * SROW + 8 * hh);
    const u4v* vb = (const u4v*)(Vsw +
        ((size_t)b * D_ + dcol + l15) * (L_ / 2) + 8 * hh);
    v8f acc = {};
    for (int kc = halfk * 32; kc < halfk * 32 + 32; ++kc) {
        ABFrag ap, bv;
        ap.q[0] = pa[kc * 4];      // dwords kc*16 + 8*hh + {0..3}
        ap.q[1] = pa[kc * 4 + 1];  // dwords kc*16 + 8*hh + {4..7}
        bv.q[0] = vb[kc * 4];      // dwords kc*16 + 8*hh + {0..7}
        bv.q[1] = vb[kc * 4 + 1];
        acc = __builtin_amdgcn_wmma_f32_16x16x32_bf16(false, ap.v, false, bv.v,
                                                      (short)0, acc, false, false);
    }
    if (halfk) {
        #pragma unroll
        for (int r = 0; r < 8; ++r) sAcc[r * 128 + t * 32 + lane] = acc[r];
    }
    __syncthreads();
    if (!halfk) {
        #pragma unroll
        for (int r = 0; r < 8; ++r) {
            int m = r + 8 * hh;
            float o = acc[r] + sAcc[r * 128 + t * 32 + lane];
            Out[((size_t)b * L_ + q0 + m) * D_ + dcol + l15] = o;
        }
    }
}

extern "C" void kernel_launch(void* const* d_in, const int* in_sizes, int n_in,
                              void* d_out, int out_size, void* d_ws, size_t ws_size,
                              hipStream_t stream) {
    const float*         q = (const float*)d_in[0];
    const float*         k = (const float*)d_in[1];
    const float*         v = (const float*)d_in[2];
    const unsigned char* m = (const unsigned char*)d_in[3];

    float* out  = (float*)d_out;                       // [B, L, D] first
    float* attn = out + (size_t)B_ * L_ * D_;          // then [B, L, L]

    // workspace: packed bf16 K (4 MB) + fragment-swizzled bf16 V (4 MB)
    unsigned int* kbf = (unsigned int*)d_ws;
    unsigned int* vsw = kbf + (size_t)B_ * L_ * (D_ / 2);

    pack_k_kernel<<<(B_ * L_ * (D_ / 2)) / 256, 256, 0, stream>>>(k, kbf);
    pack_v_kernel<<<dim3(L_ / 32, B_), 128, 0, stream>>>(v, vsw);

    dim3 grid(L_ / 16, B_);
    size_t shmem = (size_t)16 * SROW * sizeof(float)   // padded f32 scores / packed P
                 + (size_t)8 * 128 * sizeof(float);    // split-K partials
    sdpa_wmma_kernel<<<grid, 256, shmem, stream>>>(q, kbf, vsw, m, out, attn);
}
